// TimeSeriesWeighting_30116310679938
// MI455X (gfx1250) — matrix operand order repaired and checked by reading, not compiled, not run on hardware
//
#include <hip/hip_runtime.h>

typedef __attribute__((ext_vector_type(16))) _Float16 v16h;
typedef __attribute__((ext_vector_type(8)))  _Float16 v8h;
typedef __attribute__((ext_vector_type(8)))  float    v8f;

#define L_LEN   3000
#define NBINS   1501      // rfft half-spectrum bins 0..1500
#define NBP     1504      // bins padded to multiple of 16
#define TPAD    3008      // time (K) padded to multiple of 32
#define LDSP    3016      // LDS row pitch in halfs (bank-conflict pad)
#define NROWS   12288     // 1024 * 12
#define NBATCH  1024
#define NCHAN   12
#define PATCH   75
#define NPATCH  40
#define KSTEPS  94        // 3008 / 32
#define NTILES  94        // 1504 / 16

// ---------------------------------------------------------------- basis gen
__global__ void k_basis(_Float16* __restrict__ bc, _Float16* __restrict__ bs) {
  int idx = blockIdx.x * 256 + threadIdx.x;
  if (idx >= NBP * TPAD) return;
  int n = idx / TPAD;
  int t = idx - n * TPAD;
  float c = 0.f, s = 0.f;
  if (n < NBINS && t < L_LEN) {
    int r = (n * t) % L_LEN;                       // keep angle argument small
    float ang = (float)r * (6.28318530717958647692f / (float)L_LEN);
    c = cosf(ang);
    s = sinf(ang);                                  // sign irrelevant (squared)
  }
  bc[idx] = (_Float16)c;
  bs[idx] = (_Float16)s;
}

// ---------------------------------------------------------------- zero counts
__global__ void k_zero(int* __restrict__ counts) {
  int i = blockIdx.x * 256 + threadIdx.x;
  if (i < NBATCH * NPATCH) counts[i] = 0;
}

// ---------------------------------------------------------------- DFT power via WMMA
// One block = 16 rows of x, staged to LDS as f16.  8 waves split the 94
// 16-bin column tiles.  Per tile: 94 K-steps of two v_wmma_f32_16x16x32_f16
// (cos and sin basis), then energy = Re^2 + Im^2 written to the padded
// (NROWS x NBP) energy matrix.
__global__ __launch_bounds__(256) void k_gemm(const float* __restrict__ x,
                                              const _Float16* __restrict__ bC,
                                              const _Float16* __restrict__ bS,
                                              float* __restrict__ energ) {
  __shared__ _Float16 ldsA[16 * LDSP];
  const int tid = threadIdx.x;
  const int r0  = blockIdx.x * 16;

  // stage 16 rows, f32 -> f16, zero-pad t in [3000, 3008)
  for (int i = tid; i < 16 * TPAD; i += 256) {
    int row = i / TPAD;
    int t   = i - row * TPAD;
    float v = (t < L_LEN) ? x[(size_t)(r0 + row) * L_LEN + t] : 0.f;
    ldsA[row * LDSP + t] = (_Float16)v;
  }
  __syncthreads();

  const int wave = tid >> 5;
  const int lane = tid & 31;
  const int m    = lane & 15;      // A: row in tile / B,D: column in tile
  const int half = lane >> 4;      // K-half selector

  const _Float16* aL = ldsA + m * LDSP;

  for (int kt = wave; kt < NTILES; kt += 8) {
    const int n0 = kt * 16;
    const _Float16* bcL = bC + (size_t)(n0 + m) * TPAD + half * 8;
    const _Float16* bsL = bS + (size_t)(n0 + m) * TPAD + half * 8;

    v8f accRe = {};
    v8f accIm = {};

    for (int kk = 0; kk < KSTEPS; ++kk) {
      // A fragment (16-bit A 16x32 layout): lane<16 -> K {0..7,16..23},
      // lane>=16 -> K {8..15,24..31}; K packed in half pairs per VGPR.
      const int tA0 = kk * 32 + half * 8;
      v8h alo = *(const v8h*)(aL + tA0);
      v8h ahi = *(const v8h*)(aL + tA0 + 16);
      v16h a;
#pragma unroll
      for (int q = 0; q < 8; ++q) { a[q] = alo[q]; a[q + 8] = ahi[q]; }

      // B fragment: lane holds column n0+m, K {8h..8h+7, 16+8h..16+8h+7} (same as A).
      v8h bclo = *(const v8h*)(bcL + kk * 32);
      v8h bchi = *(const v8h*)(bcL + kk * 32 + 16);
      v8h bslo = *(const v8h*)(bsL + kk * 32);
      v8h bshi = *(const v8h*)(bsL + kk * 32 + 16);
      v16h vbc, vbs;
#pragma unroll
      for (int q = 0; q < 8; ++q) { vbc[q] = bclo[q]; vbc[q + 8] = bchi[q]; vbs[q] = bslo[q]; vbs[q + 8] = bshi[q]; }

      accRe = __builtin_amdgcn_wmma_f32_16x16x32_f16(false, a, false, vbc,
                                                     (short)0, accRe, false, false);
      accIm = __builtin_amdgcn_wmma_f32_16x16x32_f16(false, a, false, vbs,
                                                     (short)0, accIm, false, false);
    }

    // D layout: VGPR i -> M = i + 8*half, N = lane&15
#pragma unroll
    for (int i = 0; i < 8; ++i) {
      float e = accRe[i] * accRe[i] + accIm[i] * accIm[i];
      int M = i + (half ? 8 : 0);
      energ[(size_t)(r0 + M) * NBP + n0 + m] = e;
    }
  }
}

// ---------------------------------------------------------------- top-6 + patch counts
// One wave per row.  Scan the conceptual 3000-bin mirrored spectrum; keep a
// per-lane descending top-6 of (value, index) with the stable-argsort
// tie-break (equal value -> larger index ranks higher).  Merge via LDS,
// lane 0 finishes and accumulates integer patch counts.
__global__ __launch_bounds__(256) void k_topk(const float* __restrict__ energ,
                                              int* __restrict__ counts) {
  __shared__ float sv[8][32 * 6];
  __shared__ int   si[8][32 * 6];
  const int wave = threadIdx.x >> 5;
  const int lane = threadIdx.x & 31;
  const int row  = blockIdx.x * 8 + wave;
  const float* E = energ + (size_t)row * NBP;

  float bv[6]; int bi[6];
#pragma unroll
  for (int q = 0; q < 6; ++q) { bv[q] = -1.0f; bi[q] = -1; }

  for (int j = lane; j < L_LEN; j += 32) {
    int jf = (j <= L_LEN - j) ? j : (L_LEN - j);
    float cv = E[jf];
    int   ci = j;
#pragma unroll
    for (int q = 0; q < 6; ++q) {
      bool g = (cv > bv[q]) || (cv == bv[q] && ci > bi[q]);
      float tv = bv[q]; int ti = bi[q];
      if (g) { bv[q] = cv; bi[q] = ci; cv = tv; ci = ti; }
    }
  }
#pragma unroll
  for (int q = 0; q < 6; ++q) {
    sv[wave][lane * 6 + q] = bv[q];
    si[wave][lane * 6 + q] = bi[q];
  }
  __syncthreads();

  if (lane == 0) {
    float fv[6]; int fi[6];
#pragma unroll
    for (int q = 0; q < 6; ++q) { fv[q] = -1.0f; fi[q] = -1; }
    for (int e2 = 0; e2 < 32 * 6; ++e2) {
      float cv = sv[wave][e2];
      int   ci = si[wave][e2];
#pragma unroll
      for (int q = 0; q < 6; ++q) {
        bool g = (cv > fv[q]) || (cv == fv[q] && ci > fi[q]);
        float tv = fv[q]; int ti = fi[q];
        if (g) { fv[q] = cv; fi[q] = ci; cv = tv; ci = ti; }
      }
    }
    // ascending top6 = reversed descending; sel = fold(top6[0,2,4])
    //                = fold(desc[5]), fold(desc[3]), fold(desc[1])
    int d[3] = { fi[5], fi[3], fi[1] };
    int Tv[3], vld[3];
#pragma unroll
    for (int s = 0; s < 3; ++s) {
      int di   = d[s];
      int selv = (di <= L_LEN - di) ? di : (L_LEN - di);
      vld[s] = (selv > 0);
      Tv[s]  = vld[s] ? (L_LEN / selv) : 1;
    }
    int bidx = row / NCHAN;
    for (int p = 0; p < NPATCH; ++p) {
      int a0 = p * PATCH, b0 = a0 + PATCH;
      int h = 0;
#pragma unroll
      for (int s = 0; s < 3; ++s) {
        int kmin = (a0 + Tv[s] - 1) / Tv[s];
        h += (vld[s] && (kmin * Tv[s] < b0)) ? 1 : 0;
      }
      if (h) atomicAdd(&counts[bidx * NPATCH + p], h);
    }
  }
}

// ---------------------------------------------------------------- scale by weight
__global__ void k_scale(const int* __restrict__ counts,
                        const float* __restrict__ wptr,
                        float* __restrict__ out) {
  int i = blockIdx.x * 256 + threadIdx.x;
  if (i < NBATCH * NPATCH) out[i] = (float)counts[i] * wptr[0];
}

// ---------------------------------------------------------------- launcher
extern "C" void kernel_launch(void* const* d_in, const int* in_sizes, int n_in,
                              void* d_out, int out_size, void* d_ws, size_t ws_size,
                              hipStream_t stream) {
  (void)in_sizes; (void)n_in; (void)out_size;

  const float* x    = (const float*)d_in[0];
  const float* wgt  = (const float*)d_in[1];
  float*       out  = (float*)d_out;
  char*        ws   = (char*)d_ws;

  const size_t basisBytes = (size_t)NBP * TPAD * sizeof(_Float16);   // 9,048,064
  const size_t energBytes = (size_t)NROWS * NBP * sizeof(float);     // 73,924,608
  (void)ws_size;  // requires ~92.2 MB of workspace

  _Float16* basisC = (_Float16*)ws;
  _Float16* basisS = (_Float16*)(ws + basisBytes);
  float*    energ  = (float*)(ws + 2 * basisBytes);
  int*      counts = (int*)(ws + 2 * basisBytes + energBytes);

  const int basisTotal = NBP * TPAD;
  k_basis<<<(basisTotal + 255) / 256, 256, 0, stream>>>(basisC, basisS);
  k_zero <<<(NBATCH * NPATCH + 255) / 256, 256, 0, stream>>>(counts);
  k_gemm <<<NROWS / 16, 256, 0, stream>>>(x, basisC, basisS, energ);
  k_topk <<<NROWS / 8, 256, 0, stream>>>(energ, counts);
  k_scale<<<(NBATCH * NPATCH + 255) / 256, 256, 0, stream>>>(counts, wgt, out);
}
